// AttentionFactorizeDotProd_31078383354063
// MI455X (gfx1250) — matrix operand order repaired and checked by reading, not compiled, not run on hardware
//
#include <hip/hip_runtime.h>

#define N_BATCH 2
#define T_DIM   16
#define S_DIM   256
#define D_DIM   1024
#define QKV_DIM 3072
#define HEADS   8
#define HD      64

typedef __attribute__((ext_vector_type(16))) __bf16 v16bf;
typedef __attribute__((ext_vector_type(8)))  float  v8f;

union Frag16 {
  v16bf v;
  uint4 q[2];
  unsigned short s[16];
};

__device__ __forceinline__ unsigned short f32_to_bf16(float f) {
  unsigned int u = __float_as_uint(f);
  return (unsigned short)((u + 0x7FFFu + ((u >> 16) & 1u)) >> 16);
}

__device__ __forceinline__ v8f wmma_bf16(const Frag16& a, const Frag16& b, v8f c) {
  return __builtin_amdgcn_wmma_f32_16x16x32_bf16(false, a.v, false, b.v,
                                                 (short)0, c, false, false);
}

// ---------------- pack / prep kernels ----------------

__global__ void cvt_f32_to_bf16_kernel(const float* __restrict__ src,
                                       unsigned short* __restrict__ dst, int n) {
  int i = blockIdx.x * blockDim.x + threadIdx.x;
  if (i < n) dst[i] = f32_to_bf16(src[i]);
}

// src: [K x Nc] fp32 row-major  ->  dst: [Nc x K] bf16 (transposed)
__global__ void transpose_f32_to_bf16_kernel(const float* __restrict__ src,
                                             unsigned short* __restrict__ dst,
                                             int K, int Nc) {
  int i = blockIdx.x * blockDim.x + threadIdx.x;
  if (i >= K * Nc) return;
  int nc = i % Nc;
  int k  = i / Nc;
  dst[(size_t)nc * K + k] = f32_to_bf16(src[i]);
}

// VsT[n,t,h][hd][s]  <-  qkv[n,t,s][2048 + h*64 + hd]
__global__ void prep_vst_kernel(const unsigned short* __restrict__ qkv,
                                unsigned short* __restrict__ vsT) {
  int idx = blockIdx.x * blockDim.x + threadIdx.x;   // 4,194,304 total
  int s  = idx & 255;
  int hd = (idx >> 8) & 63;
  int h  = (idx >> 14) & 7;
  int t  = (idx >> 17) & 15;
  int n  = idx >> 21;
  vsT[idx] = qkv[((size_t)(n * T_DIM + t) * S_DIM + s) * QKV_DIM + 2048 + h * HD + hd];
}

// VtT[n,s,h][hd][t]  <-  qkv[n,t,s][2560 + h*64 + hd]
__global__ void prep_vtt_kernel(const unsigned short* __restrict__ qkv,
                                unsigned short* __restrict__ vtT) {
  int idx = blockIdx.x * blockDim.x + threadIdx.x;   // 4,194,304 total
  int t  = idx & 15;
  int hd = (idx >> 4) & 63;
  int h  = (idx >> 10) & 7;
  int s  = (idx >> 13) & 255;
  int n  = idx >> 21;
  vtT[idx] = qkv[((size_t)(n * T_DIM + t) * S_DIM + s) * QKV_DIM + 2560 + h * HD + hd];
}

// ---------------- bf16 WMMA GEMM:  C[M,Nc] = A[M,K] * BT[Nc,K]^T ----------------
// one wave -> one 32(M) x 64(N) tile; B fragments reused across both M sub-tiles;
// 12 VMEM loads per 8 WMMAs per k-step; fp32 accumulate; bf16 OR fp32(+bias) epilogue

__global__ __launch_bounds__(256) void gemm_bf16_kernel(
    const unsigned short* __restrict__ A,    // [M x K]  bf16 row-major
    const unsigned short* __restrict__ BT,   // [Nc x K] bf16 (B transposed)
    unsigned short* __restrict__ Cbf,        // bf16 output (or nullptr)
    float* __restrict__ Cf,                  // fp32 output (or nullptr)
    const float* __restrict__ bias,          // only used with Cf
    int M, int Nc, int K) {
  const int lane = threadIdx.x & 31;
  const int wave = threadIdx.x >> 5;
  const int tiles_n = Nc >> 6;
  const int tile = blockIdx.x * 8 + wave;
  if (tile >= (M >> 5) * tiles_n) return;
  const int tm = tile / tiles_n;
  const int tn = tile - tm * tiles_n;
  const int m0 = tm << 5, n0 = tn << 6;
  const int lhalf = lane >> 4, l16 = lane & 15;

  const unsigned short* arow0 = A + (size_t)(m0 + l16) * K + (lhalf ? 16 : 0);
  const unsigned short* arow1 = arow0 + (size_t)16 * K;
  const unsigned short* brow  = BT + (size_t)(n0 + l16) * K + (lhalf ? 16 : 0);

  v8f acc[2][4] = {};
  for (int k = 0; k < K; k += 32) {
    Frag16 a0, a1;
    a0.q[0] = *(const uint4*)(arow0 + k);
    a0.q[1] = *(const uint4*)(arow0 + k + 8);
    a1.q[0] = *(const uint4*)(arow1 + k);
    a1.q[1] = *(const uint4*)(arow1 + k + 8);
    __builtin_prefetch(arow0 + k + 128, 0, 1);
    __builtin_prefetch(arow1 + k + 128, 0, 1);
#pragma unroll
    for (int t4 = 0; t4 < 4; ++t4) {
      Frag16 b;
      const unsigned short* bp = brow + (size_t)(t4 * 16) * K + k;
      b.q[0] = *(const uint4*)(bp);
      b.q[1] = *(const uint4*)(bp + 8);
      acc[0][t4] = wmma_bf16(a0, b, acc[0][t4]);
      acc[1][t4] = wmma_bf16(a1, b, acc[1][t4]);
    }
  }

#pragma unroll
  for (int mi = 0; mi < 2; ++mi) {
#pragma unroll
    for (int j = 0; j < 8; ++j) {
      const int r = m0 + mi * 16 + j + 8 * lhalf;
      const size_t base = (size_t)r * Nc + n0 + l16;
      if (Cf) {
#pragma unroll
        for (int t4 = 0; t4 < 4; ++t4)
          Cf[base + t4 * 16] = acc[mi][t4][j] + bias[n0 + l16 + t4 * 16];
      } else {
#pragma unroll
        for (int t4 = 0; t4 < 4; ++t4)
          Cbf[base + t4 * 16] = f32_to_bf16(acc[mi][t4][j]);
      }
    }
  }
}

// ---------------- spatial flash attention: 1 wave per (n,t,h, 16-query strip) ----------------

__global__ __launch_bounds__(32) void spatial_attn_kernel(
    const unsigned short* __restrict__ qkv,
    const unsigned short* __restrict__ vsT,
    unsigned short* __restrict__ attn) {
  __shared__ alignas(16) unsigned short P_lds[16][32];
  const int lane = threadIdx.x;
  const int bid = blockIdx.x;
  const int qstrip = bid & 15;
  const int h = (bid >> 4) & 7;
  const int t = (bid >> 7) & 15;
  const int n = bid >> 11;
  const int lhalf = lane >> 4;
  const int l16 = lane & 15;
  const int q0 = qstrip * 16;

  const size_t frame = (size_t)(n * T_DIM + t) * S_DIM;
  const unsigned short* qrow = qkv + (frame + q0 + l16) * QKV_DIM + h * HD;

  Frag16 qa[2];
#pragma unroll
  for (int c = 0; c < 2; ++c) {
    int abase = c * 32 + (lhalf ? 16 : 0);
    qa[c].q[0] = *(const uint4*)(qrow + abase);
    qa[c].q[1] = *(const uint4*)(qrow + abase + 8);
  }

  const unsigned short* kbase = qkv + frame * QKV_DIM + 1024 + h * HD;
  const unsigned short* vbase = vsT + (size_t)((n * T_DIM + t) * HEADS + h) * HD * S_DIM;

  v8f o0 = {}, o1 = {}, o2 = {}, o3 = {};
  float m_row[8], l_row[8];
#pragma unroll
  for (int j = 0; j < 8; ++j) { m_row[j] = -3.0e38f; l_row[j] = 0.0f; }

  for (int kb = 0; kb < S_DIM / 32; ++kb) {
    v8f st0 = {}, st1 = {};
#pragma unroll
    for (int sub = 0; sub < 2; ++sub) {
      const unsigned short* krow = kbase + (size_t)(kb * 32 + sub * 16 + l16) * QKV_DIM;
      v8f sacc = {};
#pragma unroll
      for (int c = 0; c < 2; ++c) {
        Frag16 kf;
        int bbase = c * 32 + (lhalf ? 16 : 0);
        kf.q[0] = *(const uint4*)(krow + bbase);
        kf.q[1] = *(const uint4*)(krow + bbase + 8);
        sacc = wmma_bf16(qa[c], kf, sacc);
      }
      if (sub == 0) st0 = sacc; else st1 = sacc;
    }

    __syncthreads();  // single-wave WG: lowers to s_nop, keeps LDS ordering explicit
#pragma unroll
    for (int j = 0; j < 8; ++j) {
      float s0 = st0[j] * 0.125f;
      float s1 = st1[j] * 0.125f;
      float mx = fmaxf(s0, s1);
#pragma unroll
      for (int d = 1; d < 16; d <<= 1) mx = fmaxf(mx, __shfl_xor(mx, d, 16));
      float m_new = fmaxf(m_row[j], mx);
      float p0 = __expf(s0 - m_new);
      float p1 = __expf(s1 - m_new);
      float corr = __expf(m_row[j] - m_new);
      float rs = p0 + p1;
#pragma unroll
      for (int d = 1; d < 16; d <<= 1) rs += __shfl_xor(rs, d, 16);
      l_row[j] = l_row[j] * corr + rs;
      m_row[j] = m_new;
      o0[j] *= corr; o1[j] *= corr; o2[j] *= corr; o3[j] *= corr;
      int r = j + 8 * lhalf;
      P_lds[r][l16]      = f32_to_bf16(p0);
      P_lds[r][16 + l16] = f32_to_bf16(p1);
    }
    __syncthreads();

    Frag16 pa;
    {
      int base = lhalf ? 16 : 0;
      pa.q[0] = *(const uint4*)(&P_lds[l16][base]);
      pa.q[1] = *(const uint4*)(&P_lds[l16][base + 8]);
    }
#pragma unroll
    for (int tile = 0; tile < 4; ++tile) {
      const unsigned short* vrow =
          vbase + (size_t)(tile * 16 + l16) * S_DIM + kb * 32 + (lhalf ? 16 : 0);
      Frag16 vb;
      vb.q[0] = *(const uint4*)(vrow);
      vb.q[1] = *(const uint4*)(vrow + 8);
      v8f& o = tile == 0 ? o0 : tile == 1 ? o1 : tile == 2 ? o2 : o3;
      o = wmma_bf16(pa, vb, o);
    }
  }

#pragma unroll
  for (int j = 0; j < 8; ++j) {
    float inv = 1.0f / l_row[j];
    int r = q0 + j + 8 * lhalf;
    size_t ro = (frame + r) * D_DIM + h * HD + l16;
    attn[ro]      = f32_to_bf16(o0[j] * inv);
    attn[ro + 16] = f32_to_bf16(o1[j] * inv);
    attn[ro + 32] = f32_to_bf16(o2[j] * inv);
    attn[ro + 48] = f32_to_bf16(o3[j] * inv);
  }
}

// ---------------- temporal attention: 1 wave per (n,s,h); T=16 keys, zero-padded to K=32 ----------------

__global__ __launch_bounds__(32) void temporal_attn_kernel(
    const unsigned short* __restrict__ qkv,
    const unsigned short* __restrict__ vtT,
    unsigned short* __restrict__ attn) {
  __shared__ alignas(16) unsigned short P_lds[16][32];
  const int lane = threadIdx.x;
  const int bid = blockIdx.x;
  const int h = bid & 7;
  const int s = (bid >> 3) & 255;
  const int n = bid >> 11;
  const int lhalf = lane >> 4;
  const int l16 = lane & 15;

  const size_t rowq = ((size_t)(n * T_DIM + l16) * S_DIM + s) * QKV_DIM;
  const unsigned short* qrow = qkv + rowq + 512 + h * HD;
  const unsigned short* krow = qkv + rowq + 1536 + h * HD;

  v8f st = {};
#pragma unroll
  for (int c = 0; c < 2; ++c) {
    Frag16 qa, kb;
    int abase = c * 32 + (lhalf ? 16 : 0);
    qa.q[0] = *(const uint4*)(qrow + abase);
    qa.q[1] = *(const uint4*)(qrow + abase + 8);
    int bbase = c * 32 + (lhalf ? 16 : 0);
    kb.q[0] = *(const uint4*)(krow + bbase);
    kb.q[1] = *(const uint4*)(krow + bbase + 8);
    st = wmma_bf16(qa, kb, st);
  }

#pragma unroll
  for (int j = 0; j < 8; ++j) {
    float sv = st[j] * 0.125f;
    float mx = sv;
#pragma unroll
    for (int d = 1; d < 16; d <<= 1) mx = fmaxf(mx, __shfl_xor(mx, d, 16));
    float p = __expf(sv - mx);
    float rs = p;
#pragma unroll
    for (int d = 1; d < 16; d <<= 1) rs += __shfl_xor(rs, d, 16);
    p /= rs;
    int r = j + 8 * lhalf;
    P_lds[r][l16]      = f32_to_bf16(p);
    P_lds[r][16 + l16] = 0;   // zero-pad keys 16..31
  }
  __syncthreads();

  Frag16 pa;
  {
    int base = lhalf ? 16 : 0;
    pa.q[0] = *(const uint4*)(&P_lds[l16][base]);
    pa.q[1] = *(const uint4*)(&P_lds[l16][base + 8]);
  }

  const unsigned short* vbase =
      vtT + (size_t)((n * S_DIM + s) * HEADS + h) * HD * T_DIM;

  v8f o0 = {}, o1 = {}, o2 = {}, o3 = {};
#pragma unroll
  for (int tile = 0; tile < 4; ++tile) {
    Frag16 vb;
    if (lane < 16) {  // K rows 0..15 valid; rows 16..31 (upper half) are zero
      const unsigned short* vrow = vbase + (size_t)(tile * 16 + l16) * T_DIM;
      vb.q[0] = *(const uint4*)(vrow);
      vb.q[1] = *(const uint4*)(vrow + 8);
    } else {
      vb.q[0] = make_uint4(0u, 0u, 0u, 0u);
      vb.q[1] = make_uint4(0u, 0u, 0u, 0u);
    }
    v8f& o = tile == 0 ? o0 : tile == 1 ? o1 : tile == 2 ? o2 : o3;
    o = wmma_bf16(pa, vb, o);
  }

#pragma unroll
  for (int j = 0; j < 8; ++j) {
    int tq = j + 8 * lhalf;
    size_t ro = ((size_t)(n * T_DIM + tq) * S_DIM + s) * D_DIM + 512 + h * HD + l16;
    attn[ro]      = f32_to_bf16(o0[j]);
    attn[ro + 16] = f32_to_bf16(o1[j]);
    attn[ro + 32] = f32_to_bf16(o2[j]);
    attn[ro + 48] = f32_to_bf16(o3[j]);
  }
}

// ---------------- launch ----------------

extern "C" void kernel_launch(void* const* d_in, const int* in_sizes, int n_in,
                              void* d_out, int out_size, void* d_ws, size_t ws_size,
                              hipStream_t stream) {
  (void)in_sizes; (void)n_in; (void)out_size; (void)ws_size;
  const float* x     = (const float*)d_in[0];
  const float* Wqkv  = (const float*)d_in[1];
  const float* Wproj = (const float*)d_in[2];
  const float* bproj = (const float*)d_in[3];
  float* out = (float*)d_out;

  char* ws = (char*)d_ws;
  unsigned short* Xbf    = (unsigned short*)(ws + 0);          // 16,777,216 B
  unsigned short* WqkvT  = (unsigned short*)(ws + 16777216);   //  6,291,456 B
  unsigned short* WprojT = (unsigned short*)(ws + 23068672);   //  2,097,152 B
  unsigned short* qkvbf  = (unsigned short*)(ws + 25165824);   // 50,331,648 B
  unsigned short* VsT    = (unsigned short*)(ws + 75497472);   //  8,388,608 B
  unsigned short* VtT    = (unsigned short*)(ws + 83886080);   //  8,388,608 B
  unsigned short* attn   = (unsigned short*)(ws + 92274688);   // 16,777,216 B

  const int NX = 8192 * 1024;
  cvt_f32_to_bf16_kernel<<<NX / 256, 256, 0, stream>>>(x, Xbf, NX);
  transpose_f32_to_bf16_kernel<<<(1024 * 3072) / 256, 256, 0, stream>>>(Wqkv, WqkvT, 1024, 3072);
  transpose_f32_to_bf16_kernel<<<(1024 * 1024) / 256, 256, 0, stream>>>(Wproj, WprojT, 1024, 1024);

  // qkv = X @ Wqkv : 256 x 48 tiles of 32x64, 8 waves/block
  gemm_bf16_kernel<<<(256 * 48) / 8, 256, 0, stream>>>(
      Xbf, WqkvT, qkvbf, nullptr, nullptr, 8192, 3072, 1024);

  prep_vst_kernel<<<4194304 / 256, 256, 0, stream>>>(qkvbf, VsT);
  prep_vtt_kernel<<<4194304 / 256, 256, 0, stream>>>(qkvbf, VtT);

  spatial_attn_kernel<<<4096, 32, 0, stream>>>(qkvbf, VsT, attn);
  temporal_attn_kernel<<<4096, 32, 0, stream>>>(qkvbf, VtT, attn);

  // out = attn @ Wproj + bias : 256 x 16 tiles of 32x64
  gemm_bf16_kernel<<<(256 * 16) / 8, 256, 0, stream>>>(
      attn, WprojT, nullptr, out, bproj, 8192, 1024, 1024);
}
